// GNN_14130442404049
// MI455X (gfx1250) — matrix operand order repaired and checked by reading, not compiled, not run on hardware
//
#include <hip/hip_runtime.h>
#include <hip/hip_bf16.h>

// ---------------------------------------------------------------------------
// GATv2 GNN for MI455X (gfx1250, wave32).
// Dense node transforms: TDM tensor_load_to_lds staging + v_wmma_f32_16x16x32_f16.
// Edge phase: wave-per-(edge,head) reductions + atomics. BN: block-per-column.
// ---------------------------------------------------------------------------

typedef __attribute__((ext_vector_type(16))) _Float16 v16h;
typedef __attribute__((ext_vector_type(8)))  float    v8f;
typedef __attribute__((ext_vector_type(4)))  unsigned tdm_u4;
typedef __attribute__((ext_vector_type(8)))  int      tdm_i8;
typedef __attribute__((ext_vector_type(4)))  int      tdm_i4;

#ifndef __has_builtin
#define __has_builtin(x) 0
#endif
#if __has_builtin(__builtin_amdgcn_tensor_load_to_lds) && \
    __has_builtin(__builtin_amdgcn_s_wait_tensorcnt)
#define USE_TDM 1
#else
#define USE_TDM 0
#endif

#define HC      128
#define HID     32
#define HEADS   4
#define EDIM    3
#define OUTD    10
#define NGRP    64
#define NEG     0.2f
#define BNEPS   1e-5f
#define MTILE   32     // rows per GEMM block

static __device__ __forceinline__ unsigned encf(float f) {
    unsigned u = __float_as_uint(f);
    return (u & 0x80000000u) ? ~u : (u | 0x80000000u);
}
static __device__ __forceinline__ float decf(unsigned u) {
    unsigned v = (u & 0x80000000u) ? (u & 0x7fffffffu) : ~u;
    return __uint_as_float(v);
}

// ---------------- edge encoder: ea = edge_attr @ W(3x3) + b ----------------
__global__ void k_edge_enc(const float* __restrict__ eattr,
                           const float* __restrict__ W, const float* __restrict__ b,
                           float* __restrict__ ea, int E) {
    int i = blockIdx.x * blockDim.x + threadIdx.x;
    if (i >= E * EDIM) return;
    int e = i / EDIM, j = i % EDIM;
    float v = b[j];
    #pragma unroll
    for (int k = 0; k < EDIM; ++k) v += eattr[e * EDIM + k] * W[k * EDIM + j];
    ea[i] = v;
}

// ---------------- degree + loop_attr accumulation --------------------------
__global__ void k_deg_loop(const int* __restrict__ tgt, const float* __restrict__ ea,
                           float* __restrict__ loop_attr, float* __restrict__ deg, int E) {
    int e = blockIdx.x * blockDim.x + threadIdx.x;
    if (e >= E) return;
    int t = tgt[e];
    atomicAdd(&deg[t], 1.0f);
    #pragma unroll
    for (int j = 0; j < EDIM; ++j)
        atomicAdd(&loop_attr[t * EDIM + j], ea[e * EDIM + j]);
}

__global__ void k_loop_fin(float* __restrict__ loop_attr, const float* __restrict__ deg, int N) {
    int i = blockIdx.x * blockDim.x + threadIdx.x;
    if (i >= N * EDIM) return;
    loop_attr[i] /= fmaxf(deg[i / EDIM], 1.0f);
}

// ------- f32 -> f16 weight convert WITH transpose: Wht[c*128+k] = W[k*128+c]
// Makes each lane's 16-element B fragment a contiguous 32B run (2x b128 loads).
__global__ void k_w2h_t(const float* __restrict__ W, _Float16* __restrict__ Wht) {
    int i = blockIdx.x * blockDim.x + threadIdx.x;
    if (i >= HC * HC) return;
    int k = i >> 7, c = i & 127;
    Wht[(size_t)c * HC + k] = (_Float16)W[i];
}

// ---------------- WMMA GEMM: Y(N x 128) = X(N x 128) @ W(128 x 128) + bias --
// Block = 256 threads = 8 waves; block covers a 32-row tile; wave w owns
// output columns [16w, 16w+16) and keeps two 16x16 accumulators (rows 0-15 /
// 16-31 of the tile) sharing each B fragment. K loop = 4 x wmma 16x16x32 f16.
// A tile staged via TDM (tensor_load_to_lds, raw f32) then converted in LDS.
__global__ void k_gemm_wmma(const float* __restrict__ X, const _Float16* __restrict__ Wht,
                            const float* __restrict__ bias, float* __restrict__ Y, int nrows) {
    __shared__ _Float16 As16[MTILE][HC];          // 8 KB f16 tile for fragments
#if USE_TDM
    __shared__ float    Asf[MTILE][HC];           // 16 KB raw f32 tile (TDM dest)
#endif
    const int tile = blockIdx.x;
    const int base = tile * MTILE;

#if USE_TDM
    if (threadIdx.x < 32) {
        // ---- Tensor DMA descriptor (D#): 2D tile 32x128 f32, row stride 128.
        // tensor_dim1 = remaining rows so OOB rows are returned as zeros.
        unsigned ldsoff = (unsigned)(unsigned long long)(&Asf[0][0]); // LDS byte offset (flat low 32b)
        unsigned long long ga = (unsigned long long)(const void*)(X + (size_t)base * HC);
        unsigned trows = (unsigned)(nrows - base);
        tdm_u4 g0 = {};
        g0[0] = 1u;                                        // count=1 (user descriptor)
        g0[1] = ldsoff;                                    // lds_addr
        g0[2] = (unsigned)(ga & 0xffffffffu);              // global_addr[31:0]
        g0[3] = (unsigned)((ga >> 32) & 0x01ffffffu) | (2u << 30); // addr[56:32] | type=2
        tdm_i8 g1 = {};
        g1[0] = (int)(2u << 16);                           // data_size=2 (4 bytes)
        g1[1] = (int)(128u << 16);                         // tensor_dim0 = 128 (low16)
        g1[2] = (int)((trows & 0xffffu) << 16);            // tensor_dim1 low16
        g1[3] = (int)((trows >> 16) & 0xffffu) | (int)(128u << 16); // tdim1 hi | tile_dim0=128
        g1[4] = MTILE;                                     // tile_dim1=32, tile_dim2=0
        g1[5] = HC;                                        // tensor_dim0_stride = 128 elements
        g1[6] = 0;
        g1[7] = 0;
        tdm_i4 gz4 = {};
#if defined(__clang_major__) && (__clang_major__ >= 23)
        tdm_i8 gz8 = {};
        __builtin_amdgcn_tensor_load_to_lds(g0, g1, gz4, gz4, gz8, 0);
#else
        __builtin_amdgcn_tensor_load_to_lds(g0, g1, gz4, gz4, 0);
#endif
        __builtin_amdgcn_s_wait_tensorcnt(0);
    }
    __syncthreads();
    // convert staged f32 tile -> f16 tile
    for (int i = threadIdx.x; i < MTILE * HC; i += 256) {
        int r = i >> 7, c = i & 127;
        As16[r][c] = (_Float16)Asf[r][c];
    }
#else
    for (int i = threadIdx.x; i < MTILE * HC; i += 256) {
        int r = i >> 7, c = i & 127;
        int row = base + r;
        As16[r][c] = (_Float16)(row < nrows ? X[(size_t)row * HC + c] : 0.0f);
    }
#endif
    __syncthreads();

    const int lane = threadIdx.x & 31;
    const int wave = threadIdx.x >> 5;
    const int mrow = lane & 15;
    const int hi16 = lane >> 4;             // 0 for lanes 0-15, 1 for 16-31
    const int kgrp = hi16 * 16;             // A fragment K sub-offset per lane half
    const int ncol = wave * 16 + (lane & 15);
    const v16h* __restrict__ Bcol = (const v16h*)(Wht + (size_t)ncol * HC);

    v8f acc0 = {}, acc1 = {};
    #pragma unroll
    for (int kt = 0; kt < 4; ++kt) {
        v16h a0, a1, b;
        // A 16x32 f16 fragment: lanes 0-15 rows M=0..15 with
        // elements 0..15 -> K={0..15}; lanes 16-31 K +16 (matches B).
        #pragma unroll
        for (int i = 0; i < 16; ++i) {
            int k = i + kgrp + kt * 32;
            a0[i] = As16[mrow][k];
            a1[i] = As16[16 + mrow][k];
        }
        // B 32x16 fragment: lanes 0-15 hold K=0..15, lanes 16-31 K=16..31;
        // contiguous in transposed weights -> one 32B (2x b128) load.
        b = Bcol[kt * 2 + hi16];
        acc0 = __builtin_amdgcn_wmma_f32_16x16x32_f16(false, a0, false, b,
                                                      (short)0, acc0, false, false);
        acc1 = __builtin_amdgcn_wmma_f32_16x16x32_f16(false, a1, false, b,
                                                      (short)0, acc1, false, false);
    }
    // C/D layout: VGPR r, lanes 0-15 -> M=r, lanes 16-31 -> M=8+r; N = lane&15.
    const int rowbase = base + hi16 * 8;
    const float bb = bias[ncol];
    if (base + MTILE <= nrows) {            // fast path: no per-row exec masking
        #pragma unroll
        for (int r = 0; r < 8; ++r) {
            Y[(size_t)(rowbase + r) * HC + ncol]      = acc0[r] + bb;
            Y[(size_t)(rowbase + 16 + r) * HC + ncol] = acc1[r] + bb;
        }
    } else {
        #pragma unroll
        for (int r = 0; r < 8; ++r) {
            int r0 = rowbase + r, r1 = rowbase + 16 + r;
            if (r0 < nrows) Y[(size_t)r0 * HC + ncol] = acc0[r] + bb;
            if (r1 < nrows) Y[(size_t)r1 * HC + ncol] = acc1[r] + bb;
        }
    }
}

// ---------------- attention score: one wave per (edge, head) ---------------
__global__ void k_score(const float* __restrict__ xl, const float* __restrict__ xr,
                        const float* __restrict__ ea, const float* __restrict__ loop_attr,
                        const float* __restrict__ We, const float* __restrict__ att,
                        const int* __restrict__ src, const int* __restrict__ tgt,
                        float* __restrict__ score, unsigned* __restrict__ smax,
                        int E, int N) {
    int wid  = (blockIdx.x * blockDim.x + threadIdx.x) >> 5;
    int lane = threadIdx.x & 31;
    int total = (E + N) * HEADS;
    if (wid >= total) return;
    int e = wid >> 2, hh = wid & 3;
    int s, t;
    float ev0, ev1, ev2;
    if (e < E) {
        s = src[e]; t = tgt[e];
        ev0 = ea[e * EDIM]; ev1 = ea[e * EDIM + 1]; ev2 = ea[e * EDIM + 2];
    } else {
        s = t = e - E;
        ev0 = loop_attr[s * EDIM]; ev1 = loop_attr[s * EDIM + 1]; ev2 = loop_attr[s * EDIM + 2];
    }
    int ch = hh * HID + lane;
    float ef = ev0 * We[ch] + ev1 * We[HC + ch] + ev2 * We[2 * HC + ch];
    float m = xl[(size_t)s * HC + ch] + xr[(size_t)t * HC + ch] + ef;
    m = (m > 0.0f) ? m : NEG * m;
    float v = m * att[hh * HID + lane];
    #pragma unroll
    for (int off = 16; off; off >>= 1) v += __shfl_down(v, off, 32);
    if (lane == 0) {
        score[(size_t)e * HEADS + hh] = v;
        atomicMax(&smax[t * HEADS + hh], encf(v));
    }
}

// ---------------- exp(score - max) + segment sum ---------------------------
__global__ void k_expsum(float* __restrict__ score, const unsigned* __restrict__ smax,
                         float* __restrict__ ssum, const int* __restrict__ tgt,
                         int E, int N) {
    int i = blockIdx.x * blockDim.x + threadIdx.x;
    if (i >= (E + N) * HEADS) return;
    int e = i >> 2, hh = i & 3;
    int t = (e < E) ? tgt[e] : (e - E);
    float a = __expf(score[i] - decf(smax[t * HEADS + hh]));
    score[i] = a;
    atomicAdd(&ssum[t * HEADS + hh], a);
}

// ---------------- normalized weighted scatter: z[t] += a * xl[s] -----------
__global__ void k_scatter(const float* __restrict__ score, const float* __restrict__ ssum,
                          const float* __restrict__ xl,
                          const int* __restrict__ src, const int* __restrict__ tgt,
                          float* __restrict__ z, int E, int N) {
    long long i = (long long)blockIdx.x * blockDim.x + threadIdx.x;
    if (i >= (long long)(E + N) * HC) return;
    int e  = (int)(i >> 7);
    int ch = (int)(i & 127);
    int hh = ch >> 5;
    int s, t;
    if (e < E) { s = src[e]; t = tgt[e]; } else { s = t = e - E; }
    float a = score[(size_t)e * HEADS + hh] / ssum[t * HEADS + hh];
    atomicAdd(&z[(size_t)t * HC + ch], a * xl[(size_t)s * HC + ch]);
}

// ---------------- z += gat bias (+ residual) -------------------------------
__global__ void k_bias_res(float* __restrict__ z, const float* __restrict__ bias,
                           const float* __restrict__ res, int N) {
    int i = blockIdx.x * blockDim.x + threadIdx.x;
    if (i >= N * HC) return;
    float v = z[i] + bias[i & 127];
    if (res) v += res[i];
    z[i] = v;
}

// ---------------- BN column stats (one block per channel) ------------------
__global__ void k_colstats(const float* __restrict__ z, float* __restrict__ mu,
                           float* __restrict__ var, int N) {
    __shared__ float s1[256], s2[256];
    int c = blockIdx.x;
    float a = 0.0f, b = 0.0f;
    for (int r = threadIdx.x; r < N; r += 256) {
        float v = z[(size_t)r * HC + c];
        a += v; b += v * v;
    }
    s1[threadIdx.x] = a; s2[threadIdx.x] = b;
    __syncthreads();
    for (int off = 128; off; off >>= 1) {
        if (threadIdx.x < off) {
            s1[threadIdx.x] += s1[threadIdx.x + off];
            s2[threadIdx.x] += s2[threadIdx.x + off];
        }
        __syncthreads();
    }
    if (threadIdx.x == 0) {
        float m = s1[0] / (float)N;
        mu[c]  = m;
        var[c] = s2[0] / (float)N - m * m;
    }
}

// ---------------- BN apply + ELU -------------------------------------------
__global__ void k_bn_elu(const float* __restrict__ z, const float* __restrict__ mu,
                         const float* __restrict__ var, const float* __restrict__ g,
                         const float* __restrict__ b, float* __restrict__ h, int N) {
    int i = blockIdx.x * blockDim.x + threadIdx.x;
    if (i >= N * HC) return;
    int c = i & 127;
    float v = (z[i] - mu[c]) * rsqrtf(var[c] + BNEPS) * g[c] + b[c];
    h[i] = (v > 0.0f) ? v : (__expf(v) - 1.0f);
}

// ---------------- global mean pool -----------------------------------------
__global__ void k_pool(const float* __restrict__ h, const int* __restrict__ batch,
                       float* __restrict__ pooled, float* __restrict__ cnt, int N) {
    int i = blockIdx.x * blockDim.x + threadIdx.x;
    if (i >= N * HC) return;
    int n = i >> 7, c = i & 127;
    int g = batch[n];
    atomicAdd(&pooled[(size_t)g * HC + c], h[i]);
    if (c == 0) atomicAdd(&cnt[g], 1.0f);
}

// ---------------- final linear: (G x 128) @ (128 x 10) + b -----------------
__global__ void k_final(const float* __restrict__ pooled, const float* __restrict__ cnt,
                        const float* __restrict__ W, const float* __restrict__ b,
                        float* __restrict__ out) {
    int i = blockIdx.x * blockDim.x + threadIdx.x;
    if (i >= NGRP * OUTD) return;
    int g = i / OUTD, o = i % OUTD;
    float c = fmaxf(cnt[g], 1.0f);
    float acc = b[o];
    for (int k = 0; k < HC; ++k)
        acc += (pooled[(size_t)g * HC + k] / c) * W[k * OUTD + o];
    out[i] = acc;
}

// ===========================================================================
static inline int cdiv(long long a, long long b) { return (int)((a + b - 1) / b); }

extern "C" void kernel_launch(void* const* d_in, const int* in_sizes, int n_in,
                              void* d_out, int out_size, void* d_ws, size_t ws_size,
                              hipStream_t stream) {
    const int N = in_sizes[0] / HC;        // 50000
    const int E = in_sizes[2] / EDIM;      // 400000
    const int ME = E + N;                  // edges incl. self loops

    const float* x     = (const float*)d_in[0];
    const int*   src   = (const int*)d_in[1];
    const int*   tgt   = src + E;
    const float* eattr = (const float*)d_in[2];
    const int*   batch = (const int*)d_in[3];
    // d_in[4] = dropout_prob (unused, eval mode)
    const float* encW  = (const float*)d_in[5];
    const float* encB  = (const float*)d_in[6];
    // gat param groups: Wl, bl, Wr, br, We, att, b
    const int G1 = 7, BN1 = 14, G2 = 16, BN2 = 23, G3 = 25, BN3 = 32;
    const float* linW = (const float*)d_in[34];
    const float* linB = (const float*)d_in[35];
    float* out = (float*)d_out;

    // ---- workspace carve (256B aligned regions) ----
    char* wp = (char*)d_ws;
    auto carve = [&](size_t bytes) {
        void* p = (void*)wp;
        wp += (bytes + 255) & ~(size_t)255;
        return p;
    };
    float*     ea        = (float*)carve((size_t)E * EDIM * 4);
    float*     loop_attr = (float*)carve((size_t)N * EDIM * 4);
    float*     deg       = (float*)carve((size_t)N * 4);
    _Float16*  WhL       = (_Float16*)carve((size_t)HC * HC * 2);
    _Float16*  WhR       = (_Float16*)carve((size_t)HC * HC * 2);
    float*     xl        = (float*)carve((size_t)N * HC * 4);
    float*     xr        = (float*)carve((size_t)N * HC * 4);
    float*     score     = (float*)carve((size_t)ME * HEADS * 4);
    unsigned*  smax      = (unsigned*)carve((size_t)N * HEADS * 4);
    float*     ssum      = (float*)carve((size_t)N * HEADS * 4);
    float*     z         = (float*)carve((size_t)N * HC * 4);
    float*     h0        = (float*)carve((size_t)N * HC * 4);
    float*     h1        = (float*)carve((size_t)N * HC * 4);
    float*     mu        = (float*)carve(HC * 4);
    float*     var       = (float*)carve(HC * 4);
    float*     pooled    = (float*)carve((size_t)NGRP * HC * 4);
    float*     cnt       = (float*)carve((size_t)NGRP * 4);

    const int TB = 256;

    // ---- shared pre-pass: encode edges, degree, self-loop attrs ----
    hipMemsetAsync(deg, 0, (size_t)N * 4, stream);
    hipMemsetAsync(loop_attr, 0, (size_t)N * EDIM * 4, stream);
    k_edge_enc<<<cdiv((long long)E * EDIM, TB), TB, 0, stream>>>(eattr, encW, encB, ea, E);
    k_deg_loop<<<cdiv(E, TB), TB, 0, stream>>>(tgt, ea, loop_attr, deg, E);
    k_loop_fin<<<cdiv((long long)N * EDIM, TB), TB, 0, stream>>>(loop_attr, deg, N);

    auto run_layer = [&](const float* Xin, int gb, int bnb, const float* res, float* Hout) {
        const float* Wl   = (const float*)d_in[gb + 0];
        const float* bl   = (const float*)d_in[gb + 1];
        const float* Wr   = (const float*)d_in[gb + 2];
        const float* br   = (const float*)d_in[gb + 3];
        const float* We   = (const float*)d_in[gb + 4];
        const float* att  = (const float*)d_in[gb + 5];
        const float* gbias= (const float*)d_in[gb + 6];
        const float* bng  = (const float*)d_in[bnb + 0];
        const float* bnbp = (const float*)d_in[bnb + 1];

        k_w2h_t<<<cdiv(HC * HC, TB), TB, 0, stream>>>(Wl, WhL);
        k_w2h_t<<<cdiv(HC * HC, TB), TB, 0, stream>>>(Wr, WhR);
        k_gemm_wmma<<<cdiv(N, MTILE), TB, 0, stream>>>(Xin, WhL, bl, xl, N);
        k_gemm_wmma<<<cdiv(N, MTILE), TB, 0, stream>>>(Xin, WhR, br, xr, N);

        hipMemsetAsync(smax, 0, (size_t)N * HEADS * 4, stream);   // 0 encodes -max
        hipMemsetAsync(ssum, 0, (size_t)N * HEADS * 4, stream);
        hipMemsetAsync(z,    0, (size_t)N * HC * 4, stream);

        k_score<<<cdiv((long long)ME * HEADS * 32, TB), TB, 0, stream>>>(
            xl, xr, ea, loop_attr, We, att, src, tgt, score, smax, E, N);
        k_expsum<<<cdiv((long long)ME * HEADS, TB), TB, 0, stream>>>(
            score, smax, ssum, tgt, E, N);
        k_scatter<<<cdiv((long long)ME * HC, TB), TB, 0, stream>>>(
            score, ssum, xl, src, tgt, z, E, N);

        k_bias_res<<<cdiv((long long)N * HC, TB), TB, 0, stream>>>(z, gbias, res, N);
        k_colstats<<<HC, TB, 0, stream>>>(z, mu, var, N);
        k_bn_elu<<<cdiv((long long)N * HC, TB), TB, 0, stream>>>(z, mu, var, bng, bnbp, Hout, N);
    };

    run_layer(x,  G1, BN1, nullptr, h0);   // h = elu(bn(gat(x)))
    run_layer(h0, G2, BN2, h0,      h1);   // h = elu(bn(gat(h) + h))
    run_layer(h1, G3, BN3, h1,      h0);   // h = elu(bn(gat(h) + h))

    hipMemsetAsync(pooled, 0, (size_t)NGRP * HC * 4, stream);
    hipMemsetAsync(cnt,    0, (size_t)NGRP * 4, stream);
    k_pool<<<cdiv((long long)N * HC, TB), TB, 0, stream>>>(h0, batch, pooled, cnt, N);
    k_final<<<cdiv(NGRP * OUTD, TB), TB, 0, stream>>>(pooled, cnt, linW, linB, out);
}
